// ResidualBlock_22024592294337
// MI455X (gfx1250) — matrix operand order repaired and checked by reading, not compiled, not run on hardware
//
#include <hip/hip_runtime.h>
#include <hip/hip_bf16.h>
#include <math.h>

// ---------------- problem constants ----------------
#define BB    4
#define LL    2048
#define BL    (BB * LL)      // 8192 rows
#define DI    256            // d_inner
#define DS    16             // d_state
#define DTR   16             // dt_rank
#define NM    32             // modes
#define NPROJ 48             // dt_rank + 2*d_state
#define NCHAN (BB * DS * DI) // 16384 scan channels
#define CHUNK 256            // scan chunk length
#define NCH   (LL / CHUNK)   // 8 chunks

typedef __bf16 bf16;
typedef __attribute__((ext_vector_type(16))) __bf16 bf16x16;
typedef __attribute__((ext_vector_type(8)))  __bf16 bf16x8;
typedef __attribute__((ext_vector_type(8)))  float  floatx8;

__device__ __forceinline__ bf16 f2bf(float v) { return (bf16)v; }

// ---------------- WMMA fragment helpers (CDNA5 16x16x32 bf16, wave32) -------
// A (16x32 bf16), ISA 7.12.2: lanes 0-15 -> M=lane, K in {0..7,16..23};
// lanes 16-31 -> M=lane-16, K in {8..15,24..31}. Two contiguous 16B loads.
__device__ __forceinline__ bf16x16 load_a_bf16(const bf16* __restrict__ base,
                                               int ld, int k0) {
  const int lane = threadIdx.x & 31;
  const int half = lane >> 4;
  const int m    = lane & 15;
  const bf16* p  = base + (size_t)m * ld + k0 + half * 8;
  bf16x8 lo = *reinterpret_cast<const bf16x8*>(p);
  bf16x8 hi = *reinterpret_cast<const bf16x8*>(p + 16);
  bf16x16 a;
#pragma unroll
  for (int j = 0; j < 8; ++j) { a[j] = lo[j]; a[8 + j] = hi[j]; }
  return a;
}

// A fragment from fp32 source, logical K=16 zero-padded to 32 (delta GEMM).
__device__ __forceinline__ bf16x16 load_a_f32_k16(const float* __restrict__ base,
                                                  int ld) {
  const int lane = threadIdx.x & 31;
  const int half = lane >> 4;          // half=0 -> K 0..7 ; half=1 -> K 8..15
  const int m    = lane & 15;
  const float* p = base + (size_t)m * ld + half * 8;
  bf16x16 a;
#pragma unroll
  for (int j = 0; j < 8; ++j) { a[j] = f2bf(p[j]); a[8 + j] = (bf16)0.0f; }
  return a;
}

// ---- swizzled B-panel storage: fragment (kt,nt) = 512 contiguous bf16 ------
//   frag = i/512, lane = (i%512)/16, j = i%16, kt = frag/ntiles, nt = frag%ntiles
//   k = kt*32 + (j<8 ? (lane>>4)*8 + j : 16 + (lane>>4)*8 + j-8) ;  n = nt*16 + (lane&15)
// each lane's 16 B values are 32 contiguous LDS bytes -> 2x ds_load_b128.
__device__ __forceinline__ void swz_decode(int i, int ntiles, int& k, int& n) {
  const int frag   = i >> 9;
  const int within = i & 511;
  const int lane   = within >> 4;
  const int j      = within & 15;
  const int kt = frag / ntiles, nt = frag % ntiles;
  k = kt * 32 + ((j < 8) ? (lane >> 4) * 8 + j : 16 + (lane >> 4) * 8 + (j - 8));
  n = nt * 16 + (lane & 15);
}

__device__ __forceinline__ bf16x16 load_b_swz(const bf16* __restrict__ Blds,
                                              int frag) {
  const int lane = threadIdx.x & 31;
  const bf16* p  = Blds + ((size_t)frag << 9) + lane * 16;
  bf16x8 lo = *reinterpret_cast<const bf16x8*>(p);
  bf16x8 hi = *reinterpret_cast<const bf16x8*>(p + 8);
  bf16x16 b;
#pragma unroll
  for (int j = 0; j < 8; ++j) { b[j] = lo[j]; b[8 + j] = hi[j]; }
  return b;
}

__device__ __forceinline__ floatx8 wmma_bf16(bf16x16 a, bf16x16 b, floatx8 c) {
  return __builtin_amdgcn_wmma_f32_16x16x32_bf16(false, a, false, b,
                                                 (short)0, c, false, false);
}

// ---- gfx1250 async global->LDS panel stage (ASYNCcnt path, ISA 08 §4) ------
// GVS mode: vdst = LDS byte address, vaddr = 32-bit voffset, saddr = 64b base.
__device__ __forceinline__ void stage_panel_async(bf16* __restrict__ dst,
                                                  const bf16* __restrict__ src,
                                                  int nelem) {
  const unsigned ldsbase = (unsigned)(uintptr_t)dst;
  const unsigned long long gbase = (unsigned long long)(uintptr_t)src;
  for (int i = threadIdx.x * 8; i < nelem; i += blockDim.x * 8) {
    unsigned lds  = ldsbase + (unsigned)i * 2u;
    unsigned voff = (unsigned)i * 2u;
    asm volatile("global_load_async_to_lds_b128 %0, %1, %2"
                 :: "v"(lds), "v"(voff), "s"(gbase) : "memory");
  }
  asm volatile("s_wait_asynccnt 0" ::: "memory");
}

// ---------------- kernel 0: DFT matrices, emitted PRE-SWIZZLED --------------
// Fc panel (K=256 x N=64): n<32 -> cos(2*pi*k*n/256), n>=32 -> -sin (rfft)
// Gc panel (K=64 x N=256): k<32 -> cos(2*pi*k*d/256), k>=32 -> -sin (irfft)
__global__ void k_init_trig(bf16* __restrict__ Fc, bf16* __restrict__ Gc) {
  const float w = 6.28318530717958647693f / 256.0f;
  for (int idx = blockIdx.x * blockDim.x + threadIdx.x; idx < DI * 64;
       idx += gridDim.x * blockDim.x) {
    int d, x;
    swz_decode(idx, 4, d, x);            // k=d (0..255), n=x (0..63)
    float ang = w * (float)d * (float)(x & 31);
    Fc[idx] = f2bf((x < 32) ? __cosf(ang) : -__sinf(ang));
  }
  for (int idx = blockIdx.x * blockDim.x + threadIdx.x; idx < 64 * DI;
       idx += gridDim.x * blockDim.x) {
    int xr, d;
    swz_decode(idx, 16, xr, d);          // k=xr (0..63), n=d (0..255)
    float ang = w * (float)(xr & 31) * (float)d;
    Gc[idx] = f2bf((xr < 32) ? __cosf(ang) : -__sinf(ang));
  }
}

// ---------------- kernel 1: LayerNorm (one wave32 per row of 256) -----------
__global__ void __launch_bounds__(256)
k_layernorm(const float* __restrict__ x, const float* __restrict__ g,
            const float* __restrict__ b, float* __restrict__ xn,
            bf16* __restrict__ xnb) {
  const int wave = threadIdx.x >> 5;
  const int lane = threadIdx.x & 31;
  const int row  = blockIdx.x * 8 + wave;
  if (row >= BL) return;
  const float4* xr = reinterpret_cast<const float4*>(x + (size_t)row * DI);
  float4 v0 = xr[lane * 2 + 0];
  float4 v1 = xr[lane * 2 + 1];
  float vals[8] = {v0.x, v0.y, v0.z, v0.w, v1.x, v1.y, v1.z, v1.w};
  float s = 0.f, q = 0.f;
#pragma unroll
  for (int j = 0; j < 8; ++j) { s += vals[j]; q += vals[j] * vals[j]; }
#pragma unroll
  for (int m = 16; m >= 1; m >>= 1) {
    s += __shfl_xor(s, m, 32);
    q += __shfl_xor(q, m, 32);
  }
  const float mean = s * (1.0f / DI);
  const float var  = q * (1.0f / DI) - mean * mean;
  const float inv  = rsqrtf(var + 1e-5f);
  const int base = lane * 8;
#pragma unroll
  for (int j = 0; j < 8; ++j) {
    float r = (vals[j] - mean) * inv * g[base + j] + b[base + j];
    xn[(size_t)row * DI + base + j]  = r;
    xnb[(size_t)row * DI + base + j] = f2bf(r);
  }
}

// ---------------- kernel 2: dbc = xn @ W_xproj^T  (8192x256)x(256x48) -------
__global__ void __launch_bounds__(256)
k_proj_gemm(const bf16* __restrict__ xnb, const float* __restrict__ Wp,
            float* __restrict__ dbc) {
  __shared__ __align__(16) bf16 Blds[DI * NPROJ];  // 24 KB, swizzled frags
  for (int i = threadIdx.x; i < DI * NPROJ; i += blockDim.x) {
    int k, j;
    swz_decode(i, 3, k, j);              // B[k][j] = Wp[j][k]
    Blds[i] = f2bf(Wp[(size_t)j * DI + k]);
  }
  __syncthreads();
  const int wave = threadIdx.x >> 5;
  const int lane = threadIdx.x & 31;
  const int row0 = (blockIdx.x * 8 + wave) * 16;
  const bf16* Abase = xnb + (size_t)row0 * DI;
  floatx8 acc0 = {}, acc1 = {}, acc2 = {};
#pragma unroll
  for (int k0 = 0; k0 < DI; k0 += 32) {
    if (k0 + 32 < DI)   // gfx1250: global_prefetch_b8 for next A panel
      __builtin_prefetch(Abase + (size_t)(lane & 15) * DI + k0 + 32, 0, 1);
    bf16x16 a = load_a_bf16(Abase, DI, k0);
    const int kt = (k0 >> 5) * 3;
    acc0 = wmma_bf16(a, load_b_swz(Blds, kt + 0), acc0);
    acc1 = wmma_bf16(a, load_b_swz(Blds, kt + 1), acc1);
    acc2 = wmma_bf16(a, load_b_swz(Blds, kt + 2), acc2);
  }
  const int half = lane >> 4, col = lane & 15;
#pragma unroll
  for (int r = 0; r < 8; ++r) {
    int row = row0 + r + 8 * half;
    dbc[(size_t)row * NPROJ + col]      = acc0[r];
    dbc[(size_t)row * NPROJ + 16 + col] = acc1[r];
    dbc[(size_t)row * NPROJ + 32 + col] = acc2[r];
  }
}

// ---------------- kernel 3: delta = softplus(delta_r @ W_dt^T + b_dt) -------
__global__ void __launch_bounds__(256)
k_delta_gemm(const float* __restrict__ dbc, const float* __restrict__ Wdt,
             const float* __restrict__ bdt, float* __restrict__ delta) {
  __shared__ __align__(16) bf16 Blds[32 * DI];  // 16 KB, swizzled frags
  for (int i = threadIdx.x; i < 32 * DI; i += blockDim.x) {
    int k, n;
    swz_decode(i, 16, k, n);             // B[k][n] = (k<16) ? Wdt[n][k] : 0
    Blds[i] = (k < DTR) ? f2bf(Wdt[(size_t)n * DTR + k]) : (bf16)0.0f;
  }
  __syncthreads();
  const int wave = threadIdx.x >> 5;
  const int lane = threadIdx.x & 31;
  const int row0 = blockIdx.x * 16;
  bf16x16 a = load_a_f32_k16(dbc + (size_t)row0 * NPROJ, NPROJ);
  const int half = lane >> 4, col0 = lane & 15;
#pragma unroll
  for (int t = 0; t < 2; ++t) {
    const int nt = wave * 2 + t;
    floatx8 acc = {};
    acc = wmma_bf16(a, load_b_swz(Blds, nt), acc);
#pragma unroll
    for (int r = 0; r < 8; ++r) {
      int row = row0 + r + 8 * half;
      int n   = nt * 16 + col0;
      float v = acc[r] + bdt[n];
      float sp = fmaxf(v, 0.0f) + log1pf(__expf(-fabsf(v)));  // stable softplus
      delta[(size_t)row * DI + n] = sp;
    }
  }
}

// ---------------- scan: 3-phase chunked recurrence --------------------------
// phase 1: per (channel, chunk): local (prod dA, h_end) summary, h0 = 0.
// phase 2: per channel: sequential combine of NCH summaries -> chunk prefixes.
// phase 3: re-walk each chunk from its prefix state, write hs (bf16, (l,n,d)).
// Inputs re-read in phase 3 (delta+xn+dbc = 17.5 MB) are L2-resident (192 MB).
__global__ void __launch_bounds__(256)
k_scan_part(const float* __restrict__ delta, const float* __restrict__ dbc,
            const float* __restrict__ xn, const float* __restrict__ A_log,
            float* __restrict__ Asum, float* __restrict__ Hsum) {
  const int tid  = blockIdx.x * blockDim.x + threadIdx.x;  // NCHAN*NCH
  const int chan = tid & (NCHAN - 1);
  const int c    = tid >> 14;
  const int d = chan & 255, n = (chan >> 8) & 15, b = chan >> 12;
  const float A = -__expf(A_log[d * DS + n]);
  float h = 0.f, ap = 1.f;
  const size_t rowbase = (size_t)b * LL + (size_t)c * CHUNK;
  for (int l = 0; l < CHUNK; ++l) {
    const size_t row = rowbase + l;
    const float dv = delta[row * DI + d];
    const float xv = xn[row * DI + d];
    const float bc = dbc[row * NPROJ + DTR + n];
    const float da = __expf(dv * A);
    h  = da * h + dv * bc * xv;
    ap *= da;
  }
  Asum[c * NCHAN + chan] = ap;
  Hsum[c * NCHAN + chan] = h;
}

__global__ void __launch_bounds__(256)
k_scan_combine(const float* __restrict__ Asum, const float* __restrict__ Hsum,
               float* __restrict__ Hpref) {
  const int chan = blockIdx.x * blockDim.x + threadIdx.x;  // NCHAN
  float h = 0.f;
#pragma unroll
  for (int c = 0; c < NCH; ++c) {
    Hpref[c * NCHAN + chan] = h;
    h = Asum[c * NCHAN + chan] * h + Hsum[c * NCHAN + chan];
  }
}

__global__ void __launch_bounds__(256)
k_scan_fixup(const float* __restrict__ delta, const float* __restrict__ dbc,
             const float* __restrict__ xn, const float* __restrict__ A_log,
             const float* __restrict__ Hpref, bf16* __restrict__ hst) {
  const int tid  = blockIdx.x * blockDim.x + threadIdx.x;
  const int chan = tid & (NCHAN - 1);
  const int c    = tid >> 14;
  const int d = chan & 255, n = (chan >> 8) & 15, b = chan >> 12;
  const float A = -__expf(A_log[d * DS + n]);
  float h = Hpref[c * NCHAN + chan];
  const size_t rowbase = (size_t)b * LL + (size_t)c * CHUNK;
  for (int l = 0; l < CHUNK; ++l) {
    const size_t row = rowbase + l;
    const float dv = delta[row * DI + d];
    const float xv = xn[row * DI + d];
    const float bc = dbc[row * NPROJ + DTR + n];
    h = __expf(dv * A) * h + dv * bc * xv;
    hst[(row * DS + n) * DI + d] = f2bf(h);
  }
}

// ---------------- kernel 5: truncated rfft as GEMM + Cc-folded mode mix -----
// per wave: Hf(16x64) = hs_t(16x256) @ Fc(256x64), then
// P[x] = sum_i sum_o Cc[o] * ( w[i,o,x] * Hf[i,x] )   (complex, Cc folded in)
__global__ void __launch_bounds__(128)
k_fft_fwd_mix(const bf16* __restrict__ hst, const bf16* __restrict__ Fc,
              const float* __restrict__ wre, const float* __restrict__ wim,
              const float* __restrict__ dbc, bf16* __restrict__ P) {
  __shared__ __align__(16) bf16  Flds[DI * 64];   // 32 KB swizzled DFT panel
  __shared__ __align__(16) float Hf[4][16 * 64];  // 16 KB per-wave tiles
  stage_panel_async(Flds, Fc, DI * 64);
  __syncthreads();
  const int wave = threadIdx.x >> 5;
  const int lane = threadIdx.x & 31;
  const int bl   = blockIdx.x * 4 + wave;      // one (b,l) per wave
  const bf16* Abase = hst + (size_t)bl * DS * DI;
  floatx8 acc0 = {}, acc1 = {}, acc2 = {}, acc3 = {};
#pragma unroll
  for (int k0 = 0; k0 < DI; k0 += 32) {
    if (k0 + 32 < DI)
      __builtin_prefetch(Abase + (size_t)(lane & 15) * DI + k0 + 32, 0, 1);
    bf16x16 a = load_a_bf16(Abase, DI, k0);
    const int kt = (k0 >> 5) * 4;
    acc0 = wmma_bf16(a, load_b_swz(Flds, kt + 0), acc0);
    acc1 = wmma_bf16(a, load_b_swz(Flds, kt + 1), acc1);
    acc2 = wmma_bf16(a, load_b_swz(Flds, kt + 2), acc2);
    acc3 = wmma_bf16(a, load_b_swz(Flds, kt + 3), acc3);
  }
  const int half = lane >> 4, col = lane & 15;
  float* hf = Hf[wave];
#pragma unroll
  for (int r = 0; r < 8; ++r) {
    const int row = r + 8 * half;
    hf[row * 64 + col]      = acc0[r];
    hf[row * 64 + 16 + col] = acc1[r];
    hf[row * 64 + 32 + col] = acc2[r];
    hf[row * 64 + 48 + col] = acc3[r];
  }
  __syncthreads();
  float cc[16];
#pragma unroll
  for (int o = 0; o < 16; ++o) cc[o] = dbc[(size_t)bl * NPROJ + 32 + o];
  const int x = lane;  // mode 0..31
  float pre = 0.f, pim = 0.f;
#pragma unroll 4
  for (int i = 0; i < 16; ++i) {
    const float hr = hf[i * 64 + x];
    const float hi = hf[i * 64 + 32 + x];
#pragma unroll
    for (int o = 0; o < 16; ++o) {
      const float wr = wre[(i * DS + o) * NM + x];
      const float wi = wim[(i * DS + o) * NM + x];
      pre += cc[o] * (wr * hr - wi * hi);
      pim += cc[o] * (wr * hi + wi * hr);
    }
  }
  // irfft hermitian weights folded here: x=0 -> 1/256, x>0 -> 2/256
  const float s = ((x == 0) ? 1.0f : 2.0f) * (1.0f / 256.0f);
  P[(size_t)bl * 64 + x]      = f2bf(pre * s);
  P[(size_t)bl * 64 + 32 + x] = f2bf(pim * s);
}

// ---------------- kernel 6: irfft as GEMM + residual epilogue ---------------
// y = P(8192x64) @ Gc(64x256) + D*xn + x
__global__ void __launch_bounds__(256)
k_irfft_out(const bf16* __restrict__ P, const bf16* __restrict__ Gc,
            const float* __restrict__ xn, const float* __restrict__ xin,
            const float* __restrict__ Dp, float* __restrict__ out) {
  __shared__ __align__(16) bf16 Glds[64 * DI];  // 32 KB swizzled panel
  stage_panel_async(Glds, Gc, 64 * DI);
  __syncthreads();
  const int wave = threadIdx.x >> 5;
  const int lane = threadIdx.x & 31;
  const int row0 = (blockIdx.x * 8 + wave) * 16;
  bf16x16 a0 = load_a_bf16(P + (size_t)row0 * 64, 64, 0);
  bf16x16 a1 = load_a_bf16(P + (size_t)row0 * 64, 64, 32);
  const int half = lane >> 4, col0 = lane & 15;
#pragma unroll
  for (int nt = 0; nt < 16; ++nt) {
    floatx8 acc = {};
    acc = wmma_bf16(a0, load_b_swz(Glds, nt), acc);       // kt=0
    acc = wmma_bf16(a1, load_b_swz(Glds, 16 + nt), acc);  // kt=1
    const int dcol = nt * 16 + col0;
#pragma unroll
    for (int r = 0; r < 8; ++r) {
      const int row = row0 + r + 8 * half;
      const size_t idx = (size_t)row * DI + dcol;
      out[idx] = acc[r] + Dp[dcol] * xn[idx] + xin[idx];
    }
  }
}

// ---------------- host launcher ---------------------------------------------
extern "C" void kernel_launch(void* const* d_in, const int* in_sizes, int n_in,
                              void* d_out, int out_size, void* d_ws,
                              size_t ws_size, hipStream_t stream) {
  const float* x       = (const float*)d_in[0];
  const float* ln_g    = (const float*)d_in[1];
  const float* ln_b    = (const float*)d_in[2];
  const float* W_xproj = (const float*)d_in[3];
  const float* W_dt    = (const float*)d_in[4];
  const float* b_dt    = (const float*)d_in[5];
  const float* A_log   = (const float*)d_in[6];
  const float* D_param = (const float*)d_in[7];
  const float* w_re    = (const float*)d_in[8];
  const float* w_im    = (const float*)d_in[9];
  float* out = (float*)d_out;

  char* ws = (char*)d_ws;
  size_t off = 0;
  auto take = [&](size_t bytes) {
    char* p = ws + off;
    off = (off + bytes + 255) & ~(size_t)255;
    return p;
  };
  float* xn    = (float*)take((size_t)BL * DI * 4);
  bf16*  xnb   = (bf16*) take((size_t)BL * DI * 2);
  float* dbc   = (float*)take((size_t)BL * NPROJ * 4);
  float* delta = (float*)take((size_t)BL * DI * 4);
  bf16*  hst   = (bf16*) take((size_t)BL * DS * DI * 2);
  bf16*  P     = (bf16*) take((size_t)BL * 64 * 2);
  bf16*  Fc    = (bf16*) take((size_t)DI * 64 * 2);
  bf16*  Gc    = (bf16*) take((size_t)64 * DI * 2);
  float* Asum  = (float*)take((size_t)NCHAN * NCH * 4);
  float* Hsum  = (float*)take((size_t)NCHAN * NCH * 4);
  float* Hpref = (float*)take((size_t)NCHAN * NCH * 4);
  (void)ws_size; (void)n_in; (void)in_sizes; (void)out_size;

  k_init_trig   <<<32,                 256, 0, stream>>>(Fc, Gc);
  k_layernorm   <<<BL / 8,             256, 0, stream>>>(x, ln_g, ln_b, xn, xnb);
  k_proj_gemm   <<<BL / 128,           256, 0, stream>>>(xnb, W_xproj, dbc);
  k_delta_gemm  <<<BL / 16,            256, 0, stream>>>(dbc, W_dt, b_dt, delta);
  k_scan_part   <<<NCHAN * NCH / 256,  256, 0, stream>>>(delta, dbc, xn, A_log,
                                                         Asum, Hsum);
  k_scan_combine<<<NCHAN / 256,        256, 0, stream>>>(Asum, Hsum, Hpref);
  k_scan_fixup  <<<NCHAN * NCH / 256,  256, 0, stream>>>(delta, dbc, xn, A_log,
                                                         Hpref, hst);
  k_fft_fwd_mix <<<BL / 4,             128, 0, stream>>>(hst, Fc, w_re, w_im,
                                                         dbc, P);
  k_irfft_out   <<<BL / 128,           256, 0, stream>>>(P, Gc, xn, x,
                                                         D_param, out);
}
